// EMA_orig_fast_25640954757407
// MI455X (gfx1250) — matrix operand order repaired and checked by reading, not compiled, not run on hardware
//
#include <hip/hip_runtime.h>

typedef __bf16 bf16_t;
typedef __attribute__((ext_vector_type(16))) __bf16 v16bf;
typedef __attribute__((ext_vector_type(8)))  float  v8f;

#define NT   512
#define CG   16
#define HH   64
#define WW   64
#define HWSZ 4096
#define PADW 66
#define EPSV 1e-5f

// LDS layout (bytes)
#define XS_ELEMS (PADW*PADW*CG)              // 69696 bf16 (padded gx tile, channel-last)
#define XS_OFF   0
#define X2_OFF   (XS_ELEMS*2)                // 139392 : bf16 [64][64][16] conv output
#define FL_OFF   (X2_OFF + HH*WW*CG*2)       // 270464 : float scratch region
// floats: xh 1024 | xw 1024 | gH 1024 | gW 1024 | redS 512 | redQ 512
//         A1c 16 | B1c 16 | a1s 16 | a2s 16 | x2sum 16 | sigS 4096  = 9296
#define FL_FLOATS 9296
#define SMEM_BYTES (FL_OFF + FL_FLOATS*4)    // 307648 <= 327680 (320KB)

__global__ __launch_bounds__(NT)
void ema_fused_kernel(const float* __restrict__ x,
                      const float* __restrict__ w1, const float* __restrict__ b1,
                      const float* __restrict__ w3, const float* __restrict__ b3,
                      const float* __restrict__ gnw, const float* __restrict__ gnb,
                      float* __restrict__ out)
{
    extern __shared__ char smem[];
    bf16_t* xs   = (bf16_t*)(smem + XS_OFF);     // [66][66][16] bf16, halo = 0
    bf16_t* x2b  = (bf16_t*)(smem + X2_OFF);     // [64][64][16] bf16
    float*  fb   = (float*)(smem + FL_OFF);
    float*  xh    = fb;          // [16][64] row means
    float*  xw    = fb + 1024;   // [16][64] col means
    float*  gH    = fb + 2048;   // [16][64] sigmoid gate (rows)
    float*  gW    = fb + 3072;   // [16][64] sigmoid gate (cols)
    float*  redS  = fb + 4096;   // [16][32]
    float*  redQ  = fb + 4608;   // [16][32]
    float*  A1c   = fb + 5120;   // [16] x1 = xg*A1c + B1c
    float*  B1c   = fb + 5136;
    float*  a1s   = fb + 5152;   // softmax(gn_b)
    float*  a2s   = fb + 5168;   // softmax(mean(x2))
    float*  x2sum = fb + 5184;   // [16]
    float*  sigS  = fb + 5200;   // [4096] sigmoid(weights) per pixel

    const int tid  = threadIdx.x;
    const int lane = tid & 31;
    const int wv   = tid >> 5;
    const int g    = blockIdx.x;
    const float* gx   = x   + (size_t)g * (CG*HWSZ);
    float*       gout = out + (size_t)g * (CG*HWSZ);

    // ---- Build per-lane WMMA A fragments: conv weights, 2 taps packed per K=32 ----
    // A is M(out-ch) x K, K = tapsel*16 + in-ch. Lane L<16: M=L, elem e -> K=e.
    // Lane L+16: M=L, elem e -> K=16+e. So tap = 2t + (lane>=16), ic = e.
    v16bf afrag[5];
    {
        const int M = lane & 15;
        const int ibase = (lane & 16) ? 8 : 0;
        union { v16bf v; bf16_t e[16]; } u;
        #pragma unroll
        for (int t = 0; t < 5; ++t) {
            #pragma unroll
            for (int e2 = 0; e2 < 16; ++e2) {
                int tap = 2*t + (ibase >> 3);
                int ic  = e2;
                float wvv = (tap < 9) ? w3[(M*CG + ic)*9 + tap] : 0.f;
                u.e[e2] = (bf16_t)wvv;
            }
            afrag[t] = u.v;
        }
    }

    // ---- Pass 0: zero padded tile (halo must be 0) + accumulators ----
    {
        unsigned int* z = (unsigned int*)xs;
        for (int i = tid; i < XS_ELEMS/2; i += NT) z[i] = 0u;
        if (tid < CG) x2sum[tid] = 0.f;
    }
    __syncthreads();

    // ---- Pass 1: gx -> LDS bf16 channel-last padded tile (float4 global loads) ----
    for (int i = tid; i < (CG*HWSZ)/4; i += NT) {
        int c  = i >> 10;          // channel
        int pq = i & 1023;         // pixel-group of 4
        int p0 = pq << 2;
        int y  = p0 >> 6, xc = p0 & 63;   // 4 | 64, so same row
        float4 v = *(const float4*)(gx + c*HWSZ + p0);
        bf16_t* dst = xs + ((y+1)*PADW + (xc+1))*CG + c;
        dst[0*CG] = (bf16_t)v.x;
        dst[1*CG] = (bf16_t)v.y;
        dst[2*CG] = (bf16_t)v.z;
        dst[3*CG] = (bf16_t)v.w;
    }
    __syncthreads();

    // ---- Pass 2: row means x_h[c][y], col means x_w[c][x] ----
    for (int p = tid; p < CG*HH; p += NT) {
        int c = p >> 6, y = p & 63;
        float s = 0.f;
        for (int xc = 0; xc < WW; ++xc) s += (float)xs[((y+1)*PADW + (xc+1))*CG + c];
        xh[c*HH + y] = s * (1.f/WW);
    }
    for (int p = tid; p < CG*WW; p += NT) {
        int c = p >> 6, xc = p & 63;
        float s = 0.f;
        for (int y = 0; y < HH; ++y) s += (float)xs[((y+1)*PADW + (xc+1))*CG + c];
        xw[c*WW + xc] = s * (1.f/HH);
    }
    __syncthreads();

    // ---- Pass 3: hw = w1 @ [x_h; x_w] + b1 ; gates = sigmoid ----
    for (int p = tid; p < CG*128; p += NT) {
        int o = p >> 7, s = p & 127;
        float acc = b1[o];
        #pragma unroll
        for (int i = 0; i < CG; ++i)
            acc += w1[o*CG + i] * ((s < 64) ? xh[i*HH + s] : xw[i*WW + (s-64)]);
        float sg = 1.f / (1.f + __expf(-acc));
        if (s < 64) gH[o*HH + s] = sg; else gW[o*WW + (s-64)] = sg;
    }
    __syncthreads();

    // ---- Pass 4: GroupNorm stats of xg = gx*gate (fp32 from global, L2-hot) ----
    {
        int c = tid >> 5, part = tid & 31;
        float s = 0.f, q = 0.f;
        for (int k = part; k < HWSZ; k += 32) {
            int y = k >> 6, xc = k & 63;
            float v = gx[c*HWSZ + k] * gH[c*HH + y] * gW[c*WW + xc];
            s += v; q += v*v;
        }
        redS[c*32 + part] = s; redQ[c*32 + part] = q;
    }
    __syncthreads();
    if (tid < CG) {
        float s = 0.f, q = 0.f;
        for (int i = 0; i < 32; ++i) { s += redS[tid*32+i]; q += redQ[tid*32+i]; }
        float mu  = s * (1.f/HWSZ);
        float var = fmaxf(q * (1.f/HWSZ) - mu*mu, 0.f);
        float rs  = __frsqrt_rn(var + EPSV);
        float a   = rs * gnw[tid];
        A1c[tid] = a;
        B1c[tid] = gnb[tid] - mu * a;
    }
    if (tid == 32) {   // a1 = softmax(gn_b): x1 spatial mean is exactly gn_b
        float m = gnb[0];
        for (int i = 1; i < CG; ++i) m = fmaxf(m, gnb[i]);
        float e[CG]; float den = 0.f;
        for (int i = 0; i < CG; ++i) { e[i] = __expf(gnb[i]-m); den += e[i]; }
        for (int i = 0; i < CG; ++i) a1s[i] = e[i] / den;
    }
    __syncthreads();

    // ---- Pass 5: 3x3 conv via WMMA bf16, 2 taps per K=32, 5 wmma / 16-pixel tile ----
    {
        float psum[8] = {0,0,0,0,0,0,0,0};
        const int n     = lane & 15;
        const int obase = (lane & 16) ? 8 : 0;
        for (int tile = wv*16; tile < wv*16 + 16; ++tile) {
            int y  = tile >> 2;
            int x0 = (tile & 3) << 4;
            v8f acc = {0.f,0.f,0.f,0.f,0.f,0.f,0.f,0.f};
            #pragma unroll
            for (int t = 0; t < 5; ++t) {
                const int tA = 2*t;
                const int tB = (2*t+1 > 8) ? 8 : (2*t+1);   // t=4: A high-K is zero anyway
                int tap = (lane & 16) ? tB : tA;            // lanes<16 feed K0..15, lanes>=16 K16..31
                int dy = tap / 3, dx = tap - dy*3;
                const bf16_t* src = xs + ((y + dy)*PADW + (x0 + n + dx))*CG; // contiguous 16 bf16
                v16bf bfrag = *(const v16bf*)src;
                acc = __builtin_amdgcn_wmma_f32_16x16x32_bf16(
                          false, afrag[t], false, bfrag, (short)0, acc, false, false);
            }
            bf16_t* dst = x2b + ((y*WW) + x0 + n)*CG + obase;
            #pragma unroll
            for (int r = 0; r < 8; ++r) {
                float v2 = acc[r] + b3[obase + r];   // D: elem r -> M = obase+r, N = lane&15
                dst[r] = (bf16_t)v2;
                psum[r] += v2;
            }
        }
        #pragma unroll
        for (int r = 0; r < 8; ++r) atomicAdd(&x2sum[obase + r], psum[r]);
    }
    __syncthreads();

    // ---- Pass 6: a2 = softmax(mean(x2)) ----
    if (tid == 0) {
        float v[CG]; float m = -1e30f;
        for (int i = 0; i < CG; ++i) { v[i] = x2sum[i]*(1.f/HWSZ); m = fmaxf(m, v[i]); }
        float den = 0.f;
        for (int i = 0; i < CG; ++i) { v[i] = __expf(v[i]-m); den += v[i]; }
        for (int i = 0; i < CG; ++i) a2s[i] = v[i] / den;
    }
    __syncthreads();

    // ---- Pass 7a: per-pixel gate sigS[p] = sigmoid(sum_c a1*x2 + a2*x1) (all from LDS) ----
    for (int p = tid; p < HWSZ; p += NT) {
        int y = p >> 6, xc = p & 63;
        v16bf xgv = *(const v16bf*)(xs + ((y+1)*PADW + (xc+1))*CG);
        v16bf x2v = *(const v16bf*)(x2b + p*CG);
        float s = 0.f;
        #pragma unroll
        for (int c = 0; c < CG; ++c) {
            float xg = (float)xgv[c] * gH[c*HH + y] * gW[c*WW + xc];
            float x1 = xg * A1c[c] + B1c[c];
            s += a1s[c] * (float)x2v[c] + a2s[c] * x1;
        }
        sigS[p] = 1.f / (1.f + __expf(-s));
    }
    __syncthreads();

    // ---- Pass 7b: out = gx * sigS[p]  (pure streaming, float4, L2-hot read) ----
    for (int i = tid; i < (CG*HWSZ)/4; i += NT) {
        int c  = i >> 10;
        int p0 = (i & 1023) << 2;
        float4 v = *(const float4*)(gx + c*HWSZ + p0);
        float4 o;
        o.x = v.x * sigS[p0+0];
        o.y = v.y * sigS[p0+1];
        o.z = v.z * sigS[p0+2];
        o.w = v.w * sigS[p0+3];
        *(float4*)(gout + c*HWSZ + p0) = o;
    }
}

extern "C" void kernel_launch(void* const* d_in, const int* in_sizes, int n_in,
                              void* d_out, int out_size, void* d_ws, size_t ws_size,
                              hipStream_t stream) {
    const float* x   = (const float*)d_in[0];
    const float* w1  = (const float*)d_in[1];
    const float* b1  = (const float*)d_in[2];
    const float* w3  = (const float*)d_in[3];
    const float* b3  = (const float*)d_in[4];
    const float* gnw = (const float*)d_in[5];
    const float* gnb = (const float*)d_in[6];
    float* out = (float*)d_out;

    // Allow ~300KB dynamic LDS (WGP supports 320KB/workgroup on CDNA5).
    hipFuncSetAttribute(reinterpret_cast<const void*>(ema_fused_kernel),
                        hipFuncAttributeMaxDynamicSharedMemorySize, SMEM_BYTES);

    ema_fused_kernel<<<512, NT, SMEM_BYTES, stream>>>(x, w1, b1, w3, b3, gnw, gnb, out);
}
